// Qwen3NextMTPMoE_32195074850969
// MI455X (gfx1250) — compile-verified, matrix operand order checked by reading, not run on hardware
//
#include <hip/hip_runtime.h>

#define H   2048
#define E_  64
#define I_  512
#define N_  1024
#define TOPK 8
#define ILD 520  // padded LDS row stride (bf16 elems): 1040B -> rows 4 banks apart

typedef __bf16 bf16;
typedef __attribute__((ext_vector_type(16))) bf16  v16bf;
typedef __attribute__((ext_vector_type(8)))  bf16  v8bf;
typedef __attribute__((ext_vector_type(8)))  float v8f;
typedef __attribute__((ext_vector_type(4)))  float f4;

static __device__ __forceinline__ float fast_sigmoid(float x) {
  return __builtin_amdgcn_rcpf(1.f + __expf(-x));
}
static __device__ __forceinline__ float fast_silu(float x) {
  return x * fast_sigmoid(x);
}

static __device__ __forceinline__ v8f wmma_bf16(v16bf a, v16bf b, v8f c) {
  return __builtin_amdgcn_wmma_f32_16x16x32_bf16(false, a, false, b, (short)0, c,
                                                 false, false);
}

// A/B fragment (16-bit, 16x32 / 32x16) from bf16 memory: this lane's row ptr.
// Elems 0..7 -> K = k0 + 8*half ..; elems 8..15 -> K = k0 + 16 + 8*half ..
static __device__ __forceinline__ v16bf ldfrag_bf(const bf16* rowp, int k0, int half) {
  v8bf a = *(const v8bf*)(rowp + k0 + 8 * half);
  v8bf b = *(const v8bf*)(rowp + k0 + 16 + 8 * half);
  v16bf v;
#pragma unroll
  for (int i = 0; i < 8; ++i) { v[i] = a[i]; v[8 + i] = b[i]; }
  return v;
}

// Same fragment sourced from row-major fp32 memory, converted to bf16
// (native casts -> v_cvt_pk_bf16_f32 pairs).
static __device__ __forceinline__ v16bf ldfrag_f32(const float* rowp, int k0, int half) {
  const float* p0 = rowp + k0 + 8 * half;
  const float* p1 = rowp + k0 + 16 + 8 * half;
  f4 a = *(const f4*)p0; f4 b = *(const f4*)(p0 + 4);
  f4 c = *(const f4*)p1; f4 d = *(const f4*)(p1 + 4);
  v16bf v;
  v[0]  = (bf16)a.x; v[1]  = (bf16)a.y; v[2]  = (bf16)a.z; v[3]  = (bf16)a.w;
  v[4]  = (bf16)b.x; v[5]  = (bf16)b.y; v[6]  = (bf16)b.z; v[7]  = (bf16)b.w;
  v[8]  = (bf16)c.x; v[9]  = (bf16)c.y; v[10] = (bf16)c.z; v[11] = (bf16)c.w;
  v[12] = (bf16)d.x; v[13] = (bf16)d.y; v[14] = (bf16)d.z; v[15] = (bf16)d.w;
  return v;
}

// ---------------------------------------------------------------------------
// Router: logits = x @ gate_w.T, top-8 + softmax, scatter to per-expert lists.
// Also converts x -> bf16 (xb). One block per token.
// ---------------------------------------------------------------------------
__global__ __launch_bounds__(64) void router_kernel(
    const float* __restrict__ x, const float* __restrict__ gate_w,
    bf16* __restrict__ xb, int* __restrict__ counts, int* __restrict__ tids,
    float* __restrict__ tws) {
  int nid = blockIdx.x;
  int t = threadIdx.x;
  const float* xr = x + (size_t)nid * H;

  for (int i = t; i < H; i += 64) xb[(size_t)nid * H + i] = (bf16)xr[i];

  __shared__ float lg[E_];
  const float* gr = gate_w + (size_t)t * H;
  float d = 0.f;
  for (int h2 = 0; h2 < H; h2 += 4) {
    f4 a = *(const f4*)(xr + h2);
    f4 b = *(const f4*)(gr + h2);
    d += a.x * b.x + a.y * b.y + a.z * b.z + a.w * b.w;
  }
  lg[t] = d;
  __syncthreads();

  if (t == 0) {
    float v[TOPK]; int idx[TOPK];
    for (int k = 0; k < TOPK; ++k) {
      float best = -3.4e38f; int bi = 0;
      for (int e = 0; e < E_; ++e) {
        float val = lg[e];
        if (val > best) { best = val; bi = e; }
      }
      v[k] = best; idx[k] = bi; lg[bi] = -3.4e38f;
    }
    float m = v[0], ssum = 0.f, ex[TOPK];
    for (int k = 0; k < TOPK; ++k) { ex[k] = __expf(v[k] - m); ssum += ex[k]; }
    float inv = __builtin_amdgcn_rcpf(ssum);
    for (int k = 0; k < TOPK; ++k) {
      int e = idx[k];
      int slot = atomicAdd(&counts[e], 1);
      tids[e * N_ + slot] = nid;
      tws[e * N_ + slot] = ex[k] * inv;
    }
  }
}

// ---------------------------------------------------------------------------
// Shared expert: out = sigmoid(x@g1) * ((silu(x@Wg^T) * (x@Wu^T)) @ Wd^T)
// One block (8 waves) per 16-token tile. Writes (initializes) out.
// ---------------------------------------------------------------------------
__global__ __launch_bounds__(256) void shared_expert_kernel(
    const float* __restrict__ x, const bf16* __restrict__ xb,
    const float* __restrict__ sgw, const float* __restrict__ suw,
    const float* __restrict__ sdw, const float* __restrict__ g1w,
    float* __restrict__ out) {
  int tile = blockIdx.x;
  int row0 = tile * 16;
  __shared__ __attribute__((aligned(16))) bf16 interL[16 * ILD];
  __shared__ float shg[16];
  int tid = threadIdx.x;

  if (tid < 16) {  // per-row sigmoid gate
    const float* xr = x + (size_t)(row0 + tid) * H;
    float d = 0.f;
    for (int h2 = 0; h2 < H; h2 += 4) {
      f4 a = *(const f4*)(xr + h2);
      f4 b = *(const f4*)(g1w + h2);
      d += a.x * b.x + a.y * b.y + a.z * b.z + a.w * b.w;
    }
    shg[tid] = fast_sigmoid(d);
  }
  __syncthreads();

  int lane = tid & 31, w = tid >> 5;
  int n = lane & 15, half = lane >> 4;
  const bf16* arow = xb + (size_t)(row0 + n) * H;  // A row m = lane&15

  // Phase A: one A fragment feeds 4 gate + 4 up accumulators.
  const float* bgp[4]; const float* bup[4];
#pragma unroll
  for (int j = 0; j < 4; ++j) {
    int c = w * 4 + j;  // i-chunk 0..31
    bgp[j] = sgw + (size_t)(c * 16 + n) * H;
    bup[j] = suw + (size_t)(c * 16 + n) * H;
  }
  v8f accG[4] = {}; v8f accU[4] = {};
  for (int k0 = 0; k0 < H; k0 += 32) {
    v16bf A = ldfrag_bf(arow, k0, half);
#pragma unroll
    for (int j = 0; j < 4; ++j) {
      v16bf Bg = ldfrag_f32(bgp[j], k0, half);
      accG[j] = wmma_bf16(A, Bg, accG[j]);
      v16bf Bu = ldfrag_f32(bup[j], k0, half);
      accU[j] = wmma_bf16(A, Bu, accU[j]);
    }
    if ((k0 & 127) == 0 && k0 + 128 < H) {
#pragma unroll
      for (int j = 0; j < 4; ++j) {
        __builtin_prefetch(bgp[j] + k0 + 128, 0, 1);
        __builtin_prefetch(bup[j] + k0 + 128, 0, 1);
      }
    }
  }
#pragma unroll
  for (int j = 0; j < 4; ++j) {
    int c = w * 4 + j;
#pragma unroll
    for (int r = 0; r < 8; ++r) {
      int m = r + 8 * half;
      interL[m * ILD + c * 16 + n] = (bf16)(fast_silu(accG[j][r]) * accU[j][r]);
    }
  }
  __syncthreads();

  // Phase B: inter(16x512) @ Wd^T, gated store; 4 h-chunks share each A frag.
  const bf16* airow = &interL[n * ILD];
  for (int s4 = 0; s4 < 4; ++s4) {
    const float* brp[4];
#pragma unroll
    for (int q = 0; q < 4; ++q) {
      int h0 = (w * 16 + s4 * 4 + q) * 16;
      brp[q] = sdw + (size_t)(h0 + n) * I_;
    }
    v8f acc[4] = {};
    for (int k0 = 0; k0 < I_; k0 += 32) {
      v16bf A = ldfrag_bf(airow, k0, half);
#pragma unroll
      for (int q = 0; q < 4; ++q) {
        v16bf B = ldfrag_f32(brp[q], k0, half);
        acc[q] = wmma_bf16(A, B, acc[q]);
      }
    }
#pragma unroll
    for (int q = 0; q < 4; ++q) {
      int h0 = (w * 16 + s4 * 4 + q) * 16;
#pragma unroll
      for (int r = 0; r < 8; ++r) {
        int m = r + 8 * half;
        out[(size_t)(row0 + m) * H + h0 + n] = shg[m] * acc[q][r];
      }
    }
  }
}

// ---------------------------------------------------------------------------
// Routed experts: one block per (token-tile, expert); tile-major in x so
// consecutive blocks reuse one expert's 12MB of weights in L2.
// ---------------------------------------------------------------------------
__global__ __launch_bounds__(256) void moe_expert_kernel(
    const bf16* __restrict__ xb, const float* __restrict__ gu_w,
    const float* __restrict__ dp_w, const int* __restrict__ counts,
    const int* __restrict__ tids, const float* __restrict__ tws,
    float* __restrict__ out) {
  int tile = blockIdx.x;
  int e = blockIdx.y;
  int cnt = counts[e];
  if (tile * 16 >= cnt) return;

  __shared__ __attribute__((aligned(16))) bf16 interL[16 * ILD];
  __shared__ int trow[16];
  __shared__ float wrow[16];
  int tid = threadIdx.x;
  if (tid < 16) {
    int idx = tile * 16 + tid;
    if (idx < cnt) { trow[tid] = tids[e * N_ + idx]; wrow[tid] = tws[e * N_ + idx]; }
    else           { trow[tid] = 0;                  wrow[tid] = 0.f; }
  }
  __syncthreads();

  int lane = tid & 31, w = tid >> 5;
  int n = lane & 15, half = lane >> 4;
  const bf16* arow = xb + (size_t)trow[n] * H;  // gathered A row m = lane&15
  const float* guw = gu_w + (size_t)e * (2 * I_) * H;
  const float* dpw = dp_w + (size_t)e * H * I_;

  // Phase A: gate/up GEMMs; one A fragment feeds 8 accumulators.
  const float* bgp[4]; const float* bup[4];
#pragma unroll
  for (int j = 0; j < 4; ++j) {
    int c = w * 4 + j;  // gate i-chunk 0..31
    bgp[j] = guw + (size_t)(c * 16 + n) * H;
    bup[j] = guw + (size_t)(I_ + c * 16 + n) * H;
  }
  v8f accG[4] = {}; v8f accU[4] = {};
  for (int k0 = 0; k0 < H; k0 += 32) {
    v16bf A = ldfrag_bf(arow, k0, half);
#pragma unroll
    for (int j = 0; j < 4; ++j) {
      v16bf Bg = ldfrag_f32(bgp[j], k0, half);
      accG[j] = wmma_bf16(A, Bg, accG[j]);
      v16bf Bu = ldfrag_f32(bup[j], k0, half);
      accU[j] = wmma_bf16(A, Bu, accU[j]);
    }
    if ((k0 & 127) == 0 && k0 + 128 < H) {
#pragma unroll
      for (int j = 0; j < 4; ++j) {
        __builtin_prefetch(bgp[j] + k0 + 128, 0, 1);
        __builtin_prefetch(bup[j] + k0 + 128, 0, 1);
      }
    }
  }
#pragma unroll
  for (int j = 0; j < 4; ++j) {
    int c = w * 4 + j;
#pragma unroll
    for (int r = 0; r < 8; ++r) {
      int m = r + 8 * half;
      interL[m * ILD + c * 16 + n] =
          (bf16)(fast_silu(accG[j][r]) * accU[j][r] * wrow[m]);
    }
  }
  __syncthreads();

  // Phase B: inter(16x512) @ dp_w[e]^T, atomic-accumulate into out.
  const bf16* airow = &interL[n * ILD];
  for (int s4 = 0; s4 < 4; ++s4) {
    const float* brp[4];
#pragma unroll
    for (int q = 0; q < 4; ++q) {
      int h0 = (w * 16 + s4 * 4 + q) * 16;
      brp[q] = dpw + (size_t)(h0 + n) * I_;
    }
    v8f acc[4] = {};
    for (int k0 = 0; k0 < I_; k0 += 32) {
      v16bf A = ldfrag_bf(airow, k0, half);
#pragma unroll
      for (int q = 0; q < 4; ++q) {
        v16bf B = ldfrag_f32(brp[q], k0, half);
        acc[q] = wmma_bf16(A, B, acc[q]);
      }
    }
#pragma unroll
    for (int q = 0; q < 4; ++q) {
      int h0 = (w * 16 + s4 * 4 + q) * 16;
#pragma unroll
      for (int r = 0; r < 8; ++r) {
        int m = r + 8 * half;
        atomicAdd(&out[(size_t)trow[m] * H + h0 + n], acc[q][r]);
      }
    }
  }
}

// ---------------------------------------------------------------------------
extern "C" void kernel_launch(void* const* d_in, const int* in_sizes, int n_in,
                              void* d_out, int out_size, void* d_ws, size_t ws_size,
                              hipStream_t stream) {
  (void)in_sizes; (void)n_in; (void)out_size; (void)ws_size;
  const float* h          = (const float*)d_in[0];
  const float* gate_w     = (const float*)d_in[1];
  const float* gu_w       = (const float*)d_in[2];
  const float* dp_w       = (const float*)d_in[3];
  const float* sh_gate_w  = (const float*)d_in[4];
  const float* sh_up_w    = (const float*)d_in[5];
  const float* sh_down_w  = (const float*)d_in[6];
  const float* sh_gate1_w = (const float*)d_in[7];
  float* out = (float*)d_out;

  char* ws = (char*)d_ws;
  bf16* xb = (bf16*)ws;                                 // N*H bf16 = 4 MB
  size_t off = (size_t)N_ * H * sizeof(bf16);
  int* counts = (int*)(ws + off);   off += 256;         // 64 ints (padded)
  int* tids   = (int*)(ws + off);   off += (size_t)E_ * N_ * sizeof(int);
  float* tws  = (float*)(ws + off);

  hipMemsetAsync(counts, 0, E_ * sizeof(int), stream);
  router_kernel<<<N_, 64, 0, stream>>>(h, gate_w, xb, counts, tids, tws);
  shared_expert_kernel<<<N_ / 16, 256, 0, stream>>>(h, xb, sh_gate_w, sh_up_w,
                                                    sh_down_w, sh_gate1_w, out);
  moe_expert_kernel<<<dim3(N_ / 16, E_), 256, 0, stream>>>(xb, gu_w, dp_w, counts,
                                                           tids, tws, out);
}
